// TransformerDecoderLayer_81630148428087
// MI455X (gfx1250) — matrix operand order repaired and checked by reading, not compiled, not run on hardware
//
#include <hip/hip_runtime.h>

// ---------------------------------------------------------------------------
// Types for CDNA5 WMMA
// ---------------------------------------------------------------------------
typedef __attribute__((ext_vector_type(16))) __bf16 v16bf;
typedef __attribute__((ext_vector_type(8)))  float  v8f;
typedef unsigned short u16;

union FragU {
    uint4 q[2];
    v16bf v;
};

__device__ __forceinline__ u16 f2bf(float f) {
    unsigned int u = __float_as_uint(f);
    u += 0x7FFFu + ((u >> 16) & 1u);   // round-to-nearest-even
    return (u16)(u >> 16);
}
__device__ __forceinline__ float sigf(float x) { return 1.f / (1.f + __expf(-x)); }
__device__ __forceinline__ float siluf(float x) { return x / (1.f + __expf(-x)); }

// CDNA5 async global->LDS copy (ASYNCcnt-tracked, no VGPR round trip)
__device__ __forceinline__ void async_ld_b128(unsigned ldsoff, unsigned long long gaddr) {
    asm volatile("global_load_async_to_lds_b128 %0, %1, off"
                 :: "v"(ldsoff), "v"(gaddr) : "memory");
}

// ---------------------------------------------------------------------------
// Batched bf16 WMMA GEMM:  C[z] = scale * A[z] x B[z]^T (+ bias[n])
//   A row-major [M,K] (lda);  B row-major [N,K] (ldb)  ->  C[m][n] = A.Brow(n)
//   Batch offset for X in {A,B,C}: off = (z / zInner)*sXo + (z % zInner)*sXi
//   Epilogue modes (runtime):
//     accum=1           : C[idx] += (rowsc ? rowsc[m*rstride] : 1) * val
//     Cbf (gmul==null)  : Cbf[idx] = bf16(val)
//     Cbf + gmul        : Cbf[idx] = bf16(silu(gmul[idx]) * val)   (SwiGLU)
//     default           : C[idx] = val
//   Requires M % BM == 0, N % BN == 0, K % 32 == 0, lda/ldb multiples of 8.
//   Triple-buffered LDS, 100% async-to-LDS staging; steady state keeps one
//   K-chunk in flight across the wait (async ops complete in order).
// ---------------------------------------------------------------------------
template <int BM, int BN, int TM, int TN>
__global__ __launch_bounds__(256) void gemm_wmma(
    const u16* __restrict__ A, long lda, long sAi, long sAo,
    const u16* __restrict__ Bm, long ldb, long sBi, long sBo,
    float* __restrict__ C, long ldc, long sCi, long sCo,
    const float* __restrict__ bias, long sBias,
    u16* __restrict__ Cbf, const float* __restrict__ gmul,
    const float* __restrict__ rowsc, long rstride, int accum,
    int K, float scale, int zInner)
{
    constexpr int KC = 32;
    constexpr int LDS_A = KC + 8;     // pad keeps 16B alignment + spreads banks
    constexpr int LDS_B = KC + 8;
    constexpr int WM = 16 * TM, WN = 16 * TN;
    constexpr int WAVES_M = BM / WM, WAVES_N = BN / WN;
    static_assert(WAVES_M * WAVES_N == 8, "8 wave32 waves per block");
    constexpr int AV = (BM * KC) / (256 * 8);    // async ops per thread (A)
    constexpr int BV = (BN * KC) / (256 * 8);    // async ops per thread (B)
    constexpr int OPS = AV + BV;                 // per-chunk async ops

    __shared__ __align__(16) u16 As[3][BM * LDS_A];
    __shared__ __align__(16) u16 Bs[3][BN * LDS_B];   // [n][k]

    const int z  = blockIdx.z;
    const int zi = z % zInner;
    const int zo = z / zInner;
    A  += (long)zo * sAo + (long)zi * sAi;
    Bm += (long)zo * sBo + (long)zi * sBi;
    const long cofs = (long)zo * sCo + (long)zi * sCi;
    C   += cofs;
    if (Cbf)  Cbf  += cofs;
    if (gmul) gmul += cofs;
    const float* bvec = bias ? (bias + (long)z * sBias) : nullptr;

    const int m0   = blockIdx.y * BM;
    const int n0   = blockIdx.x * BN;
    const int th   = threadIdx.x;
    const int lane = th & 31;
    const int wid  = th >> 5;
    const int wm   = wid % WAVES_M;
    const int wn   = wid / WAVES_M;
    const int hi   = lane >> 4;
    const int l15  = lane & 15;

    v8f acc[TM][TN];
#pragma unroll
    for (int i = 0; i < TM; ++i)
#pragma unroll
        for (int j = 0; j < TN; ++j)
#pragma unroll
            for (int r = 0; r < 8; ++r) acc[i][j][r] = 0.f;

    // stage one K-chunk into an LDS buffer, fully async
    auto stage = [&](int k0, u16* Asb, u16* Bsb) {
#pragma unroll
        for (int it = 0; it < AV; ++it) {
            int e8 = (it * 256 + th) * 8;
            int r = e8 >> 5, c = e8 & 31;
            async_ld_b128((unsigned)(size_t)(&Asb[r * LDS_A + c]),
                          (unsigned long long)(size_t)(A + (long)(m0 + r) * lda + k0 + c));
        }
#pragma unroll
        for (int it = 0; it < BV; ++it) {
            int e8 = (it * 256 + th) * 8;
            int r = e8 >> 5, c = e8 & 31;
            async_ld_b128((unsigned)(size_t)(&Bsb[r * LDS_B + c]),
                          (unsigned long long)(size_t)(Bm + (long)(n0 + r) * ldb + k0 + c));
        }
    };

    // consume one staged K-chunk (ISA 16-bit A/B fragment layouts)
    auto compute = [&](const u16* Asb, const u16* Bsb) {
        v16bf af[TM], bfv[TN];
#pragma unroll
        for (int i = 0; i < TM; ++i) {
            // A lane holds k = i8 + 8*((i8>=8) + hi)  -> two 16B reads
            const u16* p = &Asb[(wm * WM + i * 16 + l15) * LDS_A + 8 * hi];
            FragU f;
            f.q[0] = *(const uint4*)(p);
            f.q[1] = *(const uint4*)(p + 16);
            af[i] = f.v;
        }
#pragma unroll
        for (int j = 0; j < TN; ++j) {
            // B lane holds k = i8 + 8*((i8>=8) + hi) for column n = l15 -> two 16B reads
            const u16* p = &Bsb[(wn * WN + j * 16 + l15) * LDS_B + 8 * hi];
            FragU f;
            f.q[0] = *(const uint4*)(p);
            f.q[1] = *(const uint4*)(p + 16);
            bfv[j] = f.v;
        }
#pragma unroll
        for (int i = 0; i < TM; ++i)
#pragma unroll
            for (int j = 0; j < TN; ++j)
                acc[i][j] = __builtin_amdgcn_wmma_f32_16x16x32_bf16(
                    false, af[i], false, bfv[j], (short)0, acc[i][j], false, false);
    };

    const int nchunks = K / KC;
    stage(0, As[0], Bs[0]);
    if (nchunks > 1) stage(KC, As[1], Bs[1]);

    int bc = 0;   // buffer being computed
    int bs = 2;   // next buffer to stage into
    for (int ci = 0; ci < nchunks - 1; ++ci) {
        // chunk ci complete when <= OPS (the next chunk's ops) remain in flight
        asm volatile("s_wait_asynccnt %0" :: "i"(OPS) : "memory");
        __syncthreads();      // all waves' writes for chunk ci visible
        if (ci + 2 < nchunks) {
            stage((ci + 2) * KC, As[bs], Bs[bs]);
            bs = (bs == 2) ? 0 : bs + 1;
        }
        if (ci + 3 < nchunks) {
            int kp = (ci + 3) * KC;
            __builtin_prefetch(A  + (long)(m0 + (th & (BM - 1))) * lda + kp);
            __builtin_prefetch(Bm + (long)(n0 + (th & (BN - 1))) * ldb + kp);
        }
        compute(As[bc], Bs[bc]);
        bc = (bc == 2) ? 0 : bc + 1;
    }
    // final chunk: fully drain
    asm volatile("s_wait_asynccnt 0" ::: "memory");
    __syncthreads();
    compute(As[bc], Bs[bc]);

    // ---- epilogue: C/D layout m = r + 8*hi, n = l15 -------------------
#pragma unroll
    for (int i = 0; i < TM; ++i)
#pragma unroll
        for (int j = 0; j < TN; ++j)
#pragma unroll
            for (int r = 0; r < 8; ++r) {
                int m = m0 + wm * WM + i * 16 + 8 * hi + r;
                int n = n0 + wn * WN + j * 16 + l15;
                long idx = (long)m * ldc + n;
                float val = acc[i][j][r] * scale;
                if (bvec) val += bvec[n];
                if (accum) {
                    float w = rowsc ? rowsc[(long)m * rstride] : 1.f;
                    C[idx] += w * val;
                } else if (Cbf) {
                    if (gmul) val = siluf(gmul[idx]) * val;
                    Cbf[idx] = f2bf(val);
                } else {
                    C[idx] = val;
                }
            }
}

// ---------------------------------------------------------------------------
// Elementwise / reduction kernels
// ---------------------------------------------------------------------------
__global__ void k_f2bf(const float* __restrict__ in, u16* __restrict__ out, long n) {
    long i = (long)blockIdx.x * blockDim.x + threadIdx.x;
    if (i < n) out[i] = f2bf(in[i]);
}

// batched transpose-convert: out[b][n][k] = bf16(in[b][k][n])
__global__ void k_tconv(const float* __restrict__ in, u16* __restrict__ out,
                        long Bk, long Kk, long Nn) {
    long i = (long)blockIdx.x * blockDim.x + threadIdx.x;
    long total = Bk * Kk * Nn;
    if (i >= total) return;
    long k = i % Kk;
    long t = i / Kk;
    long n = t % Nn;
    long b = t / Nn;
    out[i] = f2bf(in[(b * Kk + k) * Nn + n]);
}

// transposed head pack: out[(h*D+d)][e] = in[h][e][d]   (out ld = E)
__global__ void k_pack_heads_t(const float* __restrict__ in, u16* __restrict__ out,
                               int NHh, int Ee, int Dd) {
    long i = (long)blockIdx.x * blockDim.x + threadIdx.x;
    long total = (long)NHh * Ee * Dd;
    if (i >= total) return;
    int e = (int)(i % Ee);
    long t = i / Ee;
    int d = (int)(t % Dd);
    int h = (int)(t / Dd);
    out[i] = f2bf(in[((long)h * Ee + e) * Dd + d]);
}

__global__ void k_conv3x3(const float* __restrict__ img, const float* __restrict__ w,
                          const float* __restrict__ bb, float* __restrict__ out,
                          int Bn, int Cout, int HW) {
    long i = (long)blockIdx.x * blockDim.x + threadIdx.x;
    long total = (long)Bn * Cout * HW * HW;
    if (i >= total) return;
    int x = (int)(i % HW);
    long t = i / HW;
    int y = (int)(t % HW); t /= HW;
    int o = (int)(t % Cout);
    int b = (int)(t / Cout);
    float acc = bb[o];
#pragma unroll
    for (int c = 0; c < 3; ++c)
#pragma unroll
        for (int ky = 0; ky < 3; ++ky)
#pragma unroll
            for (int kx = 0; kx < 3; ++kx) {
                int yy = y + ky - 1, xx = x + kx - 1;
                if (yy >= 0 && yy < HW && xx >= 0 && xx < HW)
                    acc += img[((long)(b * 3 + c) * HW + yy) * HW + xx] *
                           w[((o * 3 + c) * 3 + ky) * 3 + kx];
            }
    out[i] = acc;
}

__global__ void k_gn_partial(const float* __restrict__ x, float* __restrict__ part, long perB) {
    __shared__ float s1[256], s2[256];
    int th = threadIdx.x;
    long base = (long)blockIdx.y * perB;
    float a = 0.f, b = 0.f;
    for (long i = (long)blockIdx.x * 256 + th; i < perB; i += (long)gridDim.x * 256) {
        float v = x[base + i];
        a += v; b += v * v;
    }
    s1[th] = a; s2[th] = b; __syncthreads();
    for (int o = 128; o > 0; o >>= 1) {
        if (th < o) { s1[th] += s1[th + o]; s2[th] += s2[th + o]; }
        __syncthreads();
    }
    if (th == 0) {
        long pi = (long)blockIdx.y * gridDim.x + blockIdx.x;
        part[pi * 2 + 0] = s1[0];
        part[pi * 2 + 1] = s2[0];
    }
}

__global__ void k_gn_final(const float* __restrict__ part, float* __restrict__ mv,
                           int nblk, float inv_n) {
    __shared__ float s1[256], s2[256];
    int bb = blockIdx.x, th = threadIdx.x;
    float a = 0.f, b = 0.f;
    for (int i = th; i < nblk; i += 256) {
        a += part[((long)bb * nblk + i) * 2 + 0];
        b += part[((long)bb * nblk + i) * 2 + 1];
    }
    s1[th] = a; s2[th] = b; __syncthreads();
    for (int o = 128; o > 0; o >>= 1) {
        if (th < o) { s1[th] += s1[th + o]; s2[th] += s2[th + o]; }
        __syncthreads();
    }
    if (th == 0) {
        float mean = s1[0] * inv_n;
        float var  = s2[0] * inv_n - mean * mean;
        mv[bb * 2 + 0] = mean;
        mv[bb * 2 + 1] = var;
    }
}

// GroupNorm-normalize + exact GELU + f x f average pool; out layout [b][s][e]
__global__ void k_gn_gelu_pool(const float* __restrict__ conv, const float* __restrict__ mv,
                               const float* __restrict__ g, const float* __restrict__ bta,
                               float* __restrict__ out, int Bn, int Ee, int HW, int Ds, int f) {
    long i = (long)blockIdx.x * blockDim.x + threadIdx.x;
    long total = (long)Bn * Ds * Ds * Ee;
    if (i >= total) return;
    int e = (int)(i % Ee);
    long t = i / Ee;
    int s = (int)(t % (Ds * Ds));
    int b = (int)(t / (Ds * Ds));
    int oy = s / Ds, ox = s % Ds;
    float mean = mv[b * 2 + 0];
    float inv  = rsqrtf(mv[b * 2 + 1] + 1e-5f);
    float gg = g[e], bb2 = bta[e];
    float acc = 0.f;
    for (int iy = 0; iy < 4; ++iy)
        for (int ix = 0; ix < 4; ++ix) {
            float v = conv[((long)(b * Ee + e) * HW + (oy * f + iy)) * HW + (ox * f + ix)];
            v = (v - mean) * inv * gg + bb2;
            acc += 0.5f * v * (1.f + erff(v * 0.70710678118654752f));
        }
    out[i] = acc * (1.f / (float)(f * f));
}

// LayerNorm over width 512, optional residual add, fp32 and/or bf16 outputs
__global__ void k_layernorm(const float* __restrict__ x, const float* __restrict__ add,
                            const float* __restrict__ g, const float* __restrict__ b,
                            float* __restrict__ outf, u16* __restrict__ outh) {
    __shared__ float s[256];
    long row = blockIdx.x;
    int th = threadIdx.x;
    const float* xr = x + row * 512;
    float v0 = xr[th], v1 = xr[th + 256];
    if (add) { v0 += add[row * 512 + th]; v1 += add[row * 512 + th + 256]; }
    s[th] = v0 + v1; __syncthreads();
    for (int o = 128; o > 0; o >>= 1) { if (th < o) s[th] += s[th + o]; __syncthreads(); }
    float mean = s[0] * (1.f / 512.f); __syncthreads();
    float d0 = v0 - mean, d1 = v1 - mean;
    s[th] = d0 * d0 + d1 * d1; __syncthreads();
    for (int o = 128; o > 0; o >>= 1) { if (th < o) s[th] += s[th + o]; __syncthreads(); }
    float inv = rsqrtf(s[0] * (1.f / 512.f) + 1e-5f);
    float o0 = d0 * inv * g[th] + b[th];
    float o1 = d1 * inv * g[th + 256] + b[th + 256];
    if (outf) { outf[row * 512 + th] = o0; outf[row * 512 + th + 256] = o1; }
    if (outh) { outh[row * 512 + th] = f2bf(o0); outh[row * 512 + th + 256] = f2bf(o1); }
}

__global__ void k_rmsbase(const float* __restrict__ x, float* __restrict__ out) {
    __shared__ float s[256];
    long row = blockIdx.x;
    int th = threadIdx.x;
    float v0 = x[row * 512 + th], v1 = x[row * 512 + th + 256];
    s[th] = v0 * v0 + v1 * v1; __syncthreads();
    for (int o = 128; o > 0; o >>= 1) { if (th < o) s[th] += s[th + o]; __syncthreads(); }
    float inv = rsqrtf(s[0] * (1.f / 512.f) + 1e-6f);
    out[row * 512 + th] = v0 * inv;
    out[row * 512 + th + 256] = v1 * inv;
}

// rows of 64: out_bf = l2norm(silu(x))
__global__ void k_silu_l2norm(const float* __restrict__ x, u16* __restrict__ outh) {
    __shared__ float s[64];
    long row = blockIdx.x;
    int th = threadIdx.x;
    float v = siluf(x[row * 64 + th]);
    s[th] = v * v; __syncthreads();
    for (int o = 32; o > 0; o >>= 1) { if (th < o) s[th] += s[th + o]; __syncthreads(); }
    float n = sqrtf(s[0]);
    outh[row * 64 + th] = f2bf(v / fmaxf(n, 1e-12f));
}

// rows of 64 (= (t*8+h)): out_bf = rmsnorm(x)*gH[h]*sc
__global__ void k_attnmod(const float* __restrict__ x, const float* __restrict__ gH,
                          const float* __restrict__ sc, u16* __restrict__ outh) {
    __shared__ float s[64];
    long row = blockIdx.x;
    int th = threadIdx.x;
    int h = (int)(row & 7);
    float v = x[row * 64 + th];
    s[th] = v * v; __syncthreads();
    for (int o = 32; o > 0; o >>= 1) { if (th < o) s[th] += s[th + o]; __syncthreads(); }
    float inv = rsqrtf(s[0] * (1.f / 64.f) + 1e-6f);
    outh[row * 64 + th] = f2bf(v * inv * gH[h * 64 + th] * sc[row * 64 + th]);
}

// out[t][h*64+d] = sigmoid(b2[h][d] + sum_r in1[t][h*R+r]*w2[h][r][d])
__global__ void k_headproj_sig(const float* __restrict__ in1, const float* __restrict__ w2,
                               const float* __restrict__ b2, float* __restrict__ out,
                               int R, long n) {
    long i = (long)blockIdx.x * blockDim.x + threadIdx.x;
    if (i >= n) return;
    int d = (int)(i & 63);
    long t2 = i >> 6;
    int h = (int)(t2 & 7);
    long t = t2 >> 3;
    const float* a = in1 + t * (8L * R) + h * R;
    const float* w = w2 + ((long)h * R) * 64 + d;
    float acc = b2[h * 64 + d];
    for (int r = 0; r < R; ++r) acc += a[r] * w[(long)r * 64];
    out[i] = sigf(acc);
}

// vmod = silu(hv)*alpha + sigmoid(hbeta), written TRANSPOSED:
//   in index i = ((b*1024+s)*8 + h)*64 + d   ->  out[((b*8+h)*64+d)*1024 + s]
__global__ void k_vmod_t(const float* __restrict__ hv, const float* __restrict__ alpha,
                         const float* __restrict__ hbeta, u16* __restrict__ outT, long n) {
    long i = (long)blockIdx.x * blockDim.x + threadIdx.x;
    if (i >= n) return;
    float v = siluf(hv[i]) * alpha[i] + sigf(hbeta[i]);
    int d = (int)(i & 63);
    long t2 = i >> 6;
    int h = (int)(t2 & 7);
    long tg = t2 >> 3;
    long b = tg >> 10;
    long s = tg & 1023;
    outT[(((b * 8 + h) * 64) + d) * 1024 + s] = f2bf(v);
}

// softmax over rows of 1024 -> bf16 probabilities
__global__ void k_softmax_bf(const float* __restrict__ x, u16* __restrict__ outh) {
    __shared__ float s[256];
    long row = blockIdx.x;
    int th = threadIdx.x;
    const float* xr = x + row * 1024;
    float v[4]; float m = -1e30f;
#pragma unroll
    for (int j = 0; j < 4; ++j) { v[j] = xr[th + 256 * j]; m = fmaxf(m, v[j]); }
    s[th] = m; __syncthreads();
    for (int o = 128; o > 0; o >>= 1) { if (th < o) s[th] = fmaxf(s[th], s[th + o]); __syncthreads(); }
    float mx = s[0]; __syncthreads();
    float sum = 0.f;
#pragma unroll
    for (int j = 0; j < 4; ++j) { v[j] = __expf(v[j] - mx); sum += v[j]; }
    s[th] = sum; __syncthreads();
    for (int o = 128; o > 0; o >>= 1) { if (th < o) s[th] += s[th + o]; __syncthreads(); }
    float inv = 1.f / s[0];
#pragma unroll
    for (int j = 0; j < 4; ++j) outh[row * 1024 + th + 256 * j] = f2bf(v[j] * inv);
}

// low-rank router second stage + top-4 + softmax -> combine[t][32]
__global__ void k_router(const float* __restrict__ l64, const float* __restrict__ ru,
                         float* __restrict__ comb) {
    __shared__ float lg[32];
    long t = blockIdx.x;
    int n = threadIdx.x;
    float acc = 0.f;
    for (int r = 0; r < 64; ++r) acc += l64[t * 64 + r] * ru[r * 32 + n];
    lg[n] = acc;
    comb[t * 32 + n] = 0.f;
    __syncthreads();
    if (n == 0) {
        float val[4]; int idx[4];
        unsigned mask = 0;
        for (int kk = 0; kk < 4; ++kk) {
            float bm = -1e30f; int bi = 0;
            for (int j = 0; j < 32; ++j)
                if (!((mask >> j) & 1u) && lg[j] > bm) { bm = lg[j]; bi = j; }
            val[kk] = bm; idx[kk] = bi; mask |= (1u << bi);
        }
        float mx = val[0], sum = 0.f, w[4];
        for (int kk = 0; kk < 4; ++kk) { w[kk] = __expf(val[kk] - mx); sum += w[kk]; }
        for (int kk = 0; kk < 4; ++kk) comb[t * 32 + idx[kk]] = w[kk] / sum;
    }
}

__global__ void k_scaleconv(const float* __restrict__ base, const float* __restrict__ gvec,
                            u16* __restrict__ outh, long n, int Ee) {
    long i = (long)blockIdx.x * blockDim.x + threadIdx.x;
    if (i >= n) return;
    outh[i] = f2bf(base[i] * gvec[i % Ee]);
}

__global__ void k_copy(const float* __restrict__ in, float* __restrict__ out, long n) {
    long i = (long)blockIdx.x * blockDim.x + threadIdx.x;
    if (i < n) out[i] = in[i];
}

// ---------------------------------------------------------------------------
// Host-side GEMM wrappers (B is always [N,K] row-major: C = A x B^T)
// ---------------------------------------------------------------------------
static void gemm128(hipStream_t st, const u16* A, long lda, long sAi, long sAo,
                    const u16* B, long ldb, long sBi, long sBo,
                    float* C, long ldc, long sCi, long sCo,
                    const float* bias, long sBias,
                    int M, int N, int K, float scale, int batch, int zInner,
                    u16* Cbf = nullptr, const float* gmul = nullptr,
                    const float* rowsc = nullptr, long rstride = 0, int accum = 0) {
    dim3 g(N / 128, M / 128, batch);
    gemm_wmma<128, 128, 2, 4><<<g, dim3(256), 0, st>>>(
        A, lda, sAi, sAo, B, ldb, sBi, sBo, C, ldc, sCi, sCo, bias, sBias,
        Cbf, gmul, rowsc, rstride, accum, K, scale, zInner);
}
static void gemm64(hipStream_t st, const u16* A, long lda, long sAi, long sAo,
                   const u16* B, long ldb, long sBi, long sBo,
                   float* C, long ldc, long sCi, long sCo,
                   const float* bias, long sBias,
                   int M, int N, int K, float scale, int batch, int zInner,
                   u16* Cbf = nullptr, const float* gmul = nullptr,
                   const float* rowsc = nullptr, long rstride = 0, int accum = 0) {
    dim3 g(N / 64, M / 128, batch);
    gemm_wmma<128, 64, 2, 2><<<g, dim3(256), 0, st>>>(
        A, lda, sAi, sAo, B, ldb, sBi, sBo, C, ldc, sCi, sCo, bias, sBias,
        Cbf, gmul, rowsc, rstride, accum, K, scale, zInner);
}

static inline dim3 ewg(long n) { return dim3((unsigned)((n + 255) / 256)); }

// ---------------------------------------------------------------------------
// Orchestration
// ---------------------------------------------------------------------------
extern "C" void kernel_launch(void* const* d_in, const int* in_sizes, int n_in,
                              void* d_out, int out_size, void* d_ws, size_t ws_size,
                              hipStream_t stream) {
    (void)in_sizes; (void)n_in; (void)out_size; (void)ws_size;

    const int T = 2048, E = 512, Sx = 1024, Bn = 2, NHh = 8;
    const int FS = 1536, FR = 1024, CD = 256, IMGc = 128, DFSc = 32;

    const float* image    = (const float*)d_in[0];
    const float* qin      = (const float*)d_in[1];
    const float* vin      = (const float*)d_in[2];
    const float* shortcut = (const float*)d_in[3];
    const float* qa_w = (const float*)d_in[4];  const float* qa_b = (const float*)d_in[5];
    const float* vp_w = (const float*)d_in[6];  const float* vp_b = (const float*)d_in[7];
    const float* conv_w = (const float*)d_in[8]; const float* conv_b = (const float*)d_in[9];
    const float* gn_g = (const float*)d_in[10]; const float* gn_b = (const float*)d_in[11];
    const float* ln_q_g = (const float*)d_in[12]; const float* ln_q_b = (const float*)d_in[13];
    const float* ln_k_g = (const float*)d_in[14]; const float* ln_k_b = (const float*)d_in[15];
    const float* ln_v_g = (const float*)d_in[16]; const float* ln_v_b = (const float*)d_in[17];
    const float* ln_s_g = (const float*)d_in[18]; const float* ln_s_b = (const float*)d_in[19];
    const float* ln_1_g = (const float*)d_in[20]; const float* ln_1_b = (const float*)d_in[21];
    const float* ln_2_g = (const float*)d_in[22]; const float* ln_2_b = (const float*)d_in[23];
    const float* hq_w = (const float*)d_in[24];  const float* hq_b = (const float*)d_in[25];
    const float* hk_w = (const float*)d_in[26];  const float* hk_b = (const float*)d_in[27];
    const float* hv_w = (const float*)d_in[28];  const float* hv_b = (const float*)d_in[29];
    const float* hbeta_w = (const float*)d_in[30]; const float* hbeta_b = (const float*)d_in[31];
    const float* halpha_w1 = (const float*)d_in[32]; const float* halpha_b1 = (const float*)d_in[33];
    const float* halpha_w2 = (const float*)d_in[34]; const float* halpha_b2 = (const float*)d_in[35];
    const float* hsc_w1 = (const float*)d_in[36]; const float* hsc_b1 = (const float*)d_in[37];
    const float* hsc_w2 = (const float*)d_in[38]; const float* hsc_b2 = (const float*)d_in[39];
    const float* hnorm_g = (const float*)d_in[40];
    const float* hlin1_w = (const float*)d_in[41]; const float* hlin1_b = (const float*)d_in[42];
    const float* out_w = (const float*)d_in[43]; const float* out_b = (const float*)d_in[44];
    const float* s_norm_g = (const float*)d_in[45];
    const float* s_w1 = (const float*)d_in[46]; const float* s_w3 = (const float*)d_in[47];
    const float* s_w2 = (const float*)d_in[48];
    const float* r_w1 = (const float*)d_in[49]; const float* r_w3 = (const float*)d_in[50];
    const float* r_w2 = (const float*)d_in[51];
    const float* rd_w = (const float*)d_in[52]; const float* ru_w = (const float*)d_in[53];

    // ---- workspace carving (bump allocator, 256B aligned) --------------
    char* wsb = (char*)d_ws;
    size_t off = 0;
    auto alloc = [&](size_t bytes) -> void* {
        off = (off + 255) & ~(size_t)255;
        void* p = wsb + off;
        off += bytes;
        return p;
    };
    const long TE = (long)T * E;
    // bf16 buffers (all weight copies are TRANSPOSED: [N][K])
    u16* qH   = (u16*)alloc(TE * 2);
    u16* vH   = (u16*)alloc((long)T * CD * 2);
    u16* qaW  = (u16*)alloc((long)E * E * 2);        // [512][512]
    u16* vpW  = (u16*)alloc((long)E * CD * 2);       // [512][256]
    u16* hqP  = (u16*)alloc((long)E * E * 2);        // [512(h*64+d)][512(e)]
    u16* hkP  = (u16*)alloc((long)E * E * 2);
    u16* hvP  = (u16*)alloc((long)E * E * 2);
    u16* hbP  = (u16*)alloc((long)E * E * 2);
    u16* haP  = (u16*)alloc((long)256 * E * 2);      // [256][512]
    u16* hsP  = (u16*)alloc((long)256 * E * 2);
    u16* hlinT = (u16*)alloc((long)NHh * 64 * 64 * 2); // [h][dout][din]
    u16* outW = (u16*)alloc((long)E * E * 2);
    u16* rdW  = (u16*)alloc((long)64 * E * 2);       // [64][512]
    u16* wb1  = (u16*)alloc((long)FS * E * 2);       // [FS|FR][512]
    u16* wb3  = (u16*)alloc((long)FS * E * 2);
    u16* wb2  = (u16*)alloc((long)E * FS * 2);       // [512][FS|FR]
    u16* qnH  = (u16*)alloc(TE * 2);
    u16* knH  = (u16*)alloc(TE * 2);
    u16* vnH  = (u16*)alloc(TE * 2);
    u16* scnH = (u16*)alloc(TE * 2);
    u16* qhH  = (u16*)alloc(TE * 2);
    u16* khH  = (u16*)alloc(TE * 2);
    u16* vmodT = (u16*)alloc(TE * 2);                // [(b*8+h)*64+d][s]
    u16* attnmodH = (u16*)alloc(TE * 2);
    u16* ocH  = (u16*)alloc(TE * 2);
    u16* xH   = (u16*)alloc(TE * 2);
    u16* xnH  = (u16*)alloc(TE * 2);
    u16* probsH = (u16*)alloc((long)16 * Sx * Sx * 2);
    u16* hHb  = (u16*)alloc((long)T * FS * 2);
    // fp32 buffers
    float* bigF = (float*)alloc((long)16 * Sx * Sx * 4);  // conv map, later scores
    float* partF = (float*)alloc((long)1024 * Bn * 2 * 4);
    float* mvF = (float*)alloc(4 * 4);
    float* queryF = (float*)alloc(TE * 4);
    float* kF = (float*)alloc(TE * 4);
    float* vprojF = (float*)alloc(TE * 4);
    float* qhLinF = (float*)alloc(TE * 4);
    float* khLinF = (float*)alloc(TE * 4);
    float* hvLinF = (float*)alloc(TE * 4);
    float* hbLinF = (float*)alloc(TE * 4);
    float* alpha1F = (float*)alloc((long)T * 256 * 4);
    float* sc1F = (float*)alloc((long)T * 256 * 4);
    float* alphaF = (float*)alloc(TE * 4);
    float* scF = (float*)alloc(TE * 4);
    float* attnF = (float*)alloc(TE * 4);
    float* attnOutF = (float*)alloc(TE * 4);
    float* xF = (float*)alloc(TE * 4);
    float* baseF = (float*)alloc(TE * 4);
    float* moeF = (float*)alloc(TE * 4);
    float* gF = (float*)alloc((long)T * FS * 4);
    float* l64F = (float*)alloc((long)T * 64 * 4);
    float* combF = (float*)alloc((long)T * 32 * 4);

    // ---- weight conversions / packing (all transposed to [N][K]) ------
    k_tconv<<<ewg((long)E * E), 256, 0, stream>>>(qa_w, qaW, 1, E, E);
    k_tconv<<<ewg((long)E * CD), 256, 0, stream>>>(vp_w, vpW, 1, CD, E);
    k_pack_heads_t<<<ewg((long)NHh * E * 64), 256, 0, stream>>>(hq_w, hqP, NHh, E, 64);
    k_pack_heads_t<<<ewg((long)NHh * E * 64), 256, 0, stream>>>(hk_w, hkP, NHh, E, 64);
    k_pack_heads_t<<<ewg((long)NHh * E * 64), 256, 0, stream>>>(hv_w, hvP, NHh, E, 64);
    k_pack_heads_t<<<ewg((long)NHh * E * 64), 256, 0, stream>>>(hbeta_w, hbP, NHh, E, 64);
    k_pack_heads_t<<<ewg((long)NHh * E * 32), 256, 0, stream>>>(halpha_w1, haP, NHh, E, 32);
    k_pack_heads_t<<<ewg((long)NHh * E * 32), 256, 0, stream>>>(hsc_w1, hsP, NHh, E, 32);
    k_tconv<<<ewg((long)NHh * 64 * 64), 256, 0, stream>>>(hlin1_w, hlinT, NHh, 64, 64);
    k_tconv<<<ewg((long)E * E), 256, 0, stream>>>(out_w, outW, 1, E, E);
    k_tconv<<<ewg((long)E * 64), 256, 0, stream>>>(rd_w, rdW, 1, E, 64);
    k_f2bf<<<ewg(TE), 256, 0, stream>>>(qin, qH, TE);
    k_f2bf<<<ewg((long)T * CD), 256, 0, stream>>>(vin, vH, (long)T * CD);

    // ---- query projection + LN ----------------------------------------
    gemm128(stream, qH, E, 0, 0, qaW, E, 0, 0, queryF, E, 0, 0, qa_b, 0, T, E, E, 1.f, 1, 1);
    k_layernorm<<<T, 256, 0, stream>>>(queryF, nullptr, ln_q_g, ln_q_b, nullptr, qnH);

    // ---- key path: conv -> GN -> GELU -> pool -> LN -------------------
    k_conv3x3<<<ewg((long)Bn * E * IMGc * IMGc), 256, 0, stream>>>(
        image, conv_w, conv_b, bigF, Bn, E, IMGc);
    const long perB = (long)E * IMGc * IMGc;
    k_gn_partial<<<dim3(1024, Bn), 256, 0, stream>>>(bigF, partF, perB);
    k_gn_final<<<Bn, 256, 0, stream>>>(partF, mvF, 1024, 1.f / (float)perB);
    k_gn_gelu_pool<<<ewg((long)Bn * DFSc * DFSc * E), 256, 0, stream>>>(
        bigF, mvF, gn_g, gn_b, kF, Bn, E, IMGc, DFSc, 4);
    k_layernorm<<<T, 256, 0, stream>>>(kF, nullptr, ln_k_g, ln_k_b, nullptr, knH);

    // ---- value projection + LNs ---------------------------------------
    gemm128(stream, vH, CD, 0, 0, vpW, CD, 0, 0, vprojF, E, 0, 0, vp_b, 0, T, E, CD, 1.f, 1, 1);
    k_layernorm<<<T, 256, 0, stream>>>(vprojF, nullptr, ln_v_g, ln_v_b, nullptr, vnH);
    k_layernorm<<<T, 256, 0, stream>>>(shortcut, nullptr, ln_s_g, ln_s_b, nullptr, scnH);

    // ---- per-head projections (packed GEMMs) --------------------------
    gemm128(stream, qnH, E, 0, 0, hqP, E, 0, 0, qhLinF, E, 0, 0, hq_b, 0, T, E, E, 1.f, 1, 1);
    gemm128(stream, knH, E, 0, 0, hkP, E, 0, 0, khLinF, E, 0, 0, hk_b, 0, T, E, E, 1.f, 1, 1);
    gemm128(stream, vnH, E, 0, 0, hvP, E, 0, 0, hvLinF, E, 0, 0, hv_b, 0, T, E, E, 1.f, 1, 1);
    gemm128(stream, vnH, E, 0, 0, hbP, E, 0, 0, hbLinF, E, 0, 0, hbeta_b, 0, T, E, E, 1.f, 1, 1);
    gemm128(stream, vnH, E, 0, 0, haP, E, 0, 0, alpha1F, 256, 0, 0, halpha_b1, 0, T, 256, E, 1.f, 1, 1);
    gemm128(stream, scnH, E, 0, 0, hsP, E, 0, 0, sc1F, 256, 0, 0, hsc_b1, 0, T, 256, E, 1.f, 1, 1);

    k_silu_l2norm<<<T * NHh, 64, 0, stream>>>(qhLinF, qhH);
    k_silu_l2norm<<<T * NHh, 64, 0, stream>>>(khLinF, khH);
    k_headproj_sig<<<ewg(TE), 256, 0, stream>>>(alpha1F, halpha_w2, halpha_b2, alphaF, 32, TE);
    k_headproj_sig<<<ewg(TE), 256, 0, stream>>>(sc1F, hsc_w2, hsc_b2, scF, 32, TE);
    k_vmod_t<<<ewg(TE), 256, 0, stream>>>(hvLinF, alphaF, hbLinF, vmodT, TE);

    // ---- attention: scores (Q x K^T), softmax, context ----------------
    const long SS = (long)Sx * Sx;
    gemm128(stream, qhH, E, 64, (long)Sx * E, khH, E, 64, (long)Sx * E,
            bigF, Sx, SS, 8 * SS, nullptr, 0, Sx, Sx, 64, 0.125f, 16, NHh);
    k_softmax_bf<<<16 * Sx, 256, 0, stream>>>(bigF, probsH);
    gemm64(stream, probsH, Sx, SS, 8 * SS, vmodT, Sx, 64 * (long)Sx, 8 * 64 * (long)Sx,
           attnF, E, 64, (long)Sx * E, nullptr, 0, Sx, 64, Sx, 1.f, 16, NHh);
    k_attnmod<<<T * NHh, 64, 0, stream>>>(attnF, hnorm_g, scF, attnmodH);

    // ---- per-head out linear (bf16 epilogue) + output projection ------
    gemm64(stream, attnmodH, E, 64, 0, hlinT, 64, 64 * 64, 0,
           nullptr, E, 64, 0, hlin1_b, 64, T, 64, 64, 1.f, NHh, NHh, ocH);
    gemm128(stream, ocH, E, 0, 0, outW, E, 0, 0, attnOutF, E, 0, 0, out_b, 0, T, E, E, 1.f, 1, 1);

    // ---- x = LN(residual + attn) --------------------------------------
    k_layernorm<<<T, 256, 0, stream>>>(queryF, attnOutF, ln_1_g, ln_1_b, xF, xH);
    k_rmsbase<<<T, 256, 0, stream>>>(xF, baseF);
    k_copy<<<ewg(TE), 256, 0, stream>>>(xF, moeF, TE);   // shared experts add xf

    // ---- router --------------------------------------------------------
    gemm64(stream, xH, E, 0, 0, rdW, E, 0, 0, l64F, 64, 0, 0,
           nullptr, 0, T, 64, E, 1.f, 1, 1);
    k_router<<<T, 32, 0, stream>>>(l64F, ru_w, combF);

    // ---- shared experts (8, averaged; SwiGLU + accum fused in GEMMs) --
    for (int s = 0; s < 8; ++s) {
        k_scaleconv<<<ewg(TE), 256, 0, stream>>>(baseF, s_norm_g + (long)s * E, xnH, TE, E);
        k_tconv<<<ewg((long)E * FS), 256, 0, stream>>>(s_w1 + (long)s * E * FS, wb1, 1, E, FS);
        k_tconv<<<ewg((long)E * FS), 256, 0, stream>>>(s_w3 + (long)s * E * FS, wb3, 1, E, FS);
        k_tconv<<<ewg((long)FS * E), 256, 0, stream>>>(s_w2 + (long)s * FS * E, wb2, 1, FS, E);
        gemm128(stream, xnH, E, 0, 0, wb1, E, 0, 0, gF, FS, 0, 0, nullptr, 0, T, FS, E, 1.f, 1, 1);
        gemm128(stream, xnH, E, 0, 0, wb3, E, 0, 0, nullptr, FS, 0, 0, nullptr, 0, T, FS, E, 1.f, 1, 1,
                hHb, gF);                                            // h = silu(g)*v -> bf16
        gemm128(stream, hHb, FS, 0, 0, wb2, FS, 0, 0, moeF, E, 0, 0, nullptr, 0, T, E, FS, 0.125f, 1, 1,
                nullptr, nullptr, nullptr, 0, 1);                    // moe += ffn/8
    }

    // ---- routed experts (dense scan, combine weight fused in w2 GEMM) -
    for (int e = 0; e < 32; ++e) {
        k_tconv<<<ewg((long)E * FR), 256, 0, stream>>>(r_w1 + (long)e * E * FR, wb1, 1, E, FR);
        k_tconv<<<ewg((long)E * FR), 256, 0, stream>>>(r_w3 + (long)e * E * FR, wb3, 1, E, FR);
        k_tconv<<<ewg((long)FR * E), 256, 0, stream>>>(r_w2 + (long)e * FR * E, wb2, 1, FR, E);
        gemm128(stream, xH, E, 0, 0, wb1, E, 0, 0, gF, FR, 0, 0, nullptr, 0, T, FR, E, 1.f, 1, 1);
        gemm128(stream, xH, E, 0, 0, wb3, E, 0, 0, nullptr, FR, 0, 0, nullptr, 0, T, FR, E, 1.f, 1, 1,
                hHb, gF);                                            // h = silu(g)*v -> bf16
        gemm128(stream, hHb, FR, 0, 0, wb2, FR, 0, 0, moeF, E, 0, 0, nullptr, 0, T, E, FR, 1.f, 1, 1,
                nullptr, nullptr, combF + e, 32, 1);                 // moe += cw[t]*out
    }

    // ---- out = LN(x + moe_out) ----------------------------------------
    k_layernorm<<<T, 256, 0, stream>>>(moeF, xF, ln_2_g, ln_2_b, (float*)d_out, nullptr);
}
